// mmTrans_35287451304281
// MI455X (gfx1250) — matrix operand order repaired and checked by reading, not compiled, not run on hardware
//
#include <hip/hip_runtime.h>
#include <hip/hip_bf16.h>

typedef __attribute__((ext_vector_type(16))) _Float16 v16h;
typedef __attribute__((ext_vector_type(8)))  _Float16 v8h;
typedef __attribute__((ext_vector_type(4)))  _Float16 v4h;
typedef __attribute__((ext_vector_type(8)))  float    v8f;

#define NHEAD    10
#define HEAD_DIM 50
#define DPAD     64
#define BATCH    8
#define LQDIM    2048
#define LKDIM    1024
#define PROJ     500
#define QIN      1124
#define KIN      10100

// ---- WMMA fragment loaders (contiguous k order, same for A and B) ------------------
// A 16x32 f16: lane m=lane%16,g=lane/16 holds K = 16*g + h
__device__ __forceinline__ v16h load_a_frag(const _Float16* ap, int half)
{
    const v8h lo = *(const v8h*)(ap + 16 * half);
    const v8h hi = *(const v8h*)(ap + 16 * half + 8);
    v16h a;
    #pragma unroll
    for (int i = 0; i < 8; ++i) { a[i] = lo[i]; a[i + 8] = hi[i]; }
    return a;
}
// B 32x16 f16 from B^T tile: lane n=lane%16,kg=lane/16 holds K = 16*kg + h
__device__ __forceinline__ v16h load_b_frag(const _Float16* bp)
{
    const v8h lo = *(const v8h*)bp;
    const v8h hi = *(const v8h*)(bp + 8);
    v16h b;
    #pragma unroll
    for (int i = 0; i < 8; ++i) { b[i] = lo[i]; b[i + 8] = hi[i]; }
    return b;
}

// ---------------------------------------------------------------------------
// Kernel 1: out = X(f32)[M,K] @ W(f32)[K,PROJ] + bias -> f16 padded head layout
// [b, head, l, 64].  64x64 tile / block, 128 thr = 4 waves.  Register prefetch
// + LDS ping-pong: one barrier per 32-k step; global loads of tile i+1 overlap
// the WMMAs of tile i.
// ---------------------------------------------------------------------------
template <bool KG, bool NG>
__device__ __forceinline__ void proj_load_regs(const float* __restrict__ X,
                                               const float* __restrict__ W,
                                               int K, int mtile, int ntile,
                                               int k0, int t,
                                               _Float16 aR[16], _Float16 bR[4][4])
{
    // A: thread t -> row t>>1, 16 contiguous k  (global_load_b128 x4)
    const int ai  = t >> 1;
    const int aj0 = (t & 1) * 16;
    const float* asrc = X + (size_t)(mtile + ai) * K + k0 + aj0;
    #pragma unroll
    for (int j = 0; j < 16; ++j) {
        const float v = (!KG || (k0 + aj0 + j < K)) ? asrc[j] : 0.0f;
        aR[j] = (_Float16)v;
    }
    // B: thread t -> 4k x 4n sub-block; coalesced float4 per k-row
    const int kb  = (t >> 4) * 4;        // 0,4,...,28
    const int nn0 = (t & 15) * 4;        // 0..60
    #pragma unroll
    for (int i = 0; i < 4; ++i) {
        const int krow = k0 + kb + i;
        const float* bsrc = W + (size_t)krow * PROJ + ntile + nn0;
        const bool kok = (!KG) || (krow < K);
        #pragma unroll
        for (int j = 0; j < 4; ++j) {
            const bool nok = (!NG) || (ntile + nn0 + j < PROJ);
            const float v = (kok && nok) ? bsrc[j] : 0.0f;
            bR[i][j] = (_Float16)v;
        }
    }
}

__device__ __forceinline__ void proj_store_lds(_Float16 (* __restrict__ As)[40],
                                               _Float16 (* __restrict__ Bt)[40],
                                               int t, const _Float16 aR[16],
                                               const _Float16 bR[4][4])
{
    const int ai = t >> 1, aj0 = (t & 1) * 16;
    #pragma unroll
    for (int j = 0; j < 16; ++j) As[ai][aj0 + j] = aR[j];   // 2x ds_store_b128
    const int kb = (t >> 4) * 4, nn0 = (t & 15) * 4;
    #pragma unroll
    for (int j = 0; j < 4; ++j) {                            // 4x ds_store_b64
        v4h p;
        #pragma unroll
        for (int i = 0; i < 4; ++i) p[i] = bR[i][j];
        *(v4h*)&Bt[nn0 + j][kb] = p;
    }
}

__global__ __launch_bounds__(128)
void proj_gemm_kernel(const float* __restrict__ X, const float* __restrict__ W,
                      const float* __restrict__ bias, _Float16* __restrict__ out,
                      int K, int Lrows)
{
    __shared__ __align__(16) _Float16 As[2][64][40];
    __shared__ __align__(16) _Float16 Bt[2][64][40];

    const int t      = threadIdx.x;
    const int wave   = t >> 5;
    const int lane   = t & 31;
    const int lane16 = lane & 15;
    const int half   = lane >> 4;

    const int mtile = blockIdx.x * 64;
    const int ntile = blockIdx.y * 64;
    const bool nfull = (ntile + 64 <= PROJ);      // uniform per block

    v8f acc[4];
    #pragma unroll
    for (int nb = 0; nb < 4; ++nb)
        #pragma unroll
        for (int r = 0; r < 8; ++r) acc[nb][r] = 0.0f;

    const int nk = (K + 31) / 32;
    _Float16 aR[16], bR[4][4];

    if (nfull) proj_load_regs<false, false>(X, W, K, mtile, ntile, 0, t, aR, bR);
    else       proj_load_regs<false, true >(X, W, K, mtile, ntile, 0, t, aR, bR);
    proj_store_lds(As[0], Bt[0], t, aR, bR);
    __syncthreads();

    int cur = 0;
    for (int it = 0; it < nk; ++it) {
        const bool has_next = (it + 1 < nk);
        const int  knext    = (it + 1) * 32;
        if (has_next) {                           // global prefetch of tile i+1
            if (knext + 32 <= K) {
                if (nfull) proj_load_regs<false, false>(X, W, K, mtile, ntile, knext, t, aR, bR);
                else       proj_load_regs<false, true >(X, W, K, mtile, ntile, knext, t, aR, bR);
            } else {
                if (nfull) proj_load_regs<true, false>(X, W, K, mtile, ntile, knext, t, aR, bR);
                else       proj_load_regs<true, true >(X, W, K, mtile, ntile, knext, t, aR, bR);
            }
        }

        // fragment loads as one clause, then WMMAs back-to-back
        const v16h a = load_a_frag(&As[cur][wave * 16 + lane16][0], half);
        v16h b[4];
        #pragma unroll
        for (int nb = 0; nb < 4; ++nb)
            b[nb] = load_b_frag(&Bt[cur][nb * 16 + lane16][16 * half]);
        __builtin_amdgcn_sched_barrier(0);
        #pragma unroll
        for (int nb = 0; nb < 4; ++nb)
            acc[nb] = __builtin_amdgcn_wmma_f32_16x16x32_f16(
                false, a, false, b[nb], (short)0, acc[nb], false, false);

        if (has_next)                              // stage tile i+1 to other buf
            proj_store_lds(As[cur ^ 1], Bt[cur ^ 1], t, aR, bR);
        __syncthreads();                           // one barrier per k-step
        cur ^= 1;
    }

    // D[lane][r] = C[8*half + r][lane16]; scatter into [b, head, l, 64] f16
    #pragma unroll
    for (int nb = 0; nb < 4; ++nb) {
        const int c = ntile + nb * 16 + lane16;
        if (c < PROJ) {
            const int head = c / HEAD_DIM;
            const int d    = c % HEAD_DIM;
            const float bv = bias[c];
            #pragma unroll
            for (int r = 0; r < 8; ++r) {
                const int m    = mtile + wave * 16 + half * 8 + r;
                const int bidx = m / Lrows;
                const int l    = m % Lrows;
                out[(((size_t)bidx * NHEAD + head) * Lrows + l) * DPAD + d] =
                    (_Float16)(acc[nb][r] + bv);
            }
        }
    }
}

// ---------------------------------------------------------------------------
// Kernel 2: per-(b,h) streaming-softmax attention.  K tiles double-buffered in
// LDS and staged with ASYNCcnt-tracked global_load_async_to_lds_b128: tile i+1
// is in flight while the 8 WMMAs + softmax of tile i execute.
// ---------------------------------------------------------------------------
__device__ __forceinline__ void async_stage_64x64(const _Float16* __restrict__ g0,
                                                  _Float16 (* __restrict__ dst)[72],
                                                  int t)
{
    // 64 rows x 64 halves = 512 x b128, 128 lanes x 4 each
    const int i  = t >> 1;
    const int j0 = (t & 1) * 32;
    const _Float16* g = g0 + (size_t)i * DPAD + j0;
    const unsigned lds = (unsigned)(size_t)&dst[i][j0];   // low 32b = LDS offset
    #pragma unroll
    for (int j = 0; j < 4; ++j) {
        asm volatile("global_load_async_to_lds_b128 %0, %1, off"
                     :: "v"(lds + 16u * j), "v"(g + 8 * j) : "memory");
    }
}

__global__ __launch_bounds__(128)
void attn_kernel(const _Float16* __restrict__ qproj,
                 const _Float16* __restrict__ kproj,
                 const float* __restrict__ value,
                 float* __restrict__ attn_out)
{
    __shared__ __align__(16) _Float16 Qt[64][72];
    __shared__ __align__(16) _Float16 Kt[2][64][72];
    __shared__ float vs[2][64];

    const int t      = threadIdx.x;
    const int wave   = t >> 5;
    const int lane   = t & 31;
    const int lane16 = lane & 15;
    const int half   = lane >> 4;

    const int qtile = blockIdx.x * 64;
    const int bh    = blockIdx.y;
    const int b     = bh / NHEAD;
    const int h     = bh % NHEAD;

    const _Float16* qbase = qproj + (size_t)bh * LQDIM * DPAD;
    const _Float16* kbase = kproj + (size_t)bh * LKDIM * DPAD;
    const float*    vbase = value + (size_t)b * LKDIM;

    async_stage_64x64(qbase + (size_t)qtile * DPAD, Qt, t);
    async_stage_64x64(kbase, Kt[0], t);
    if (t < 64) vs[0][t] = vbase[t];
    asm volatile("s_wait_asynccnt 0x0" ::: "memory");
    __syncthreads();

    // A fragments (2 k-steps over padded head_dim 64), loaded once
    v16h a0, a1;
    {
        const _Float16* ap = &Qt[wave * 16 + lane16][0];
        a0 = load_a_frag(ap, half);
        a1 = load_a_frag(ap + 32, half);
    }

    const float scale = 0.141421356237309515f;   // 1/sqrt(50)
    float mrow[8], srow[8], orow[8];
    #pragma unroll
    for (int r = 0; r < 8; ++r) { mrow[r] = -1e30f; srow[r] = 0.f; orow[r] = 0.f; }

    const int ntiles = LKDIM / 64;
    for (int it = 0; it < ntiles; ++it) {
        const int cur = it & 1;
        const int nxt = cur ^ 1;
        if (it + 1 < ntiles) {                    // async prefetch next K tile
            async_stage_64x64(kbase + (size_t)(it + 1) * 64 * DPAD, Kt[nxt], t);
            if (t < 64) vs[nxt][t] = vbase[(it + 1) * 64 + t];
        }

        // all 8 B fragments as one clause, then 8 WMMAs back-to-back
        v16h bfr[8];
        #pragma unroll
        for (int nb = 0; nb < 4; ++nb) {
            const _Float16* bp = &Kt[cur][nb * 16 + lane16][16 * half];
            bfr[2 * nb]     = load_b_frag(bp);
            bfr[2 * nb + 1] = load_b_frag(bp + 32);
        }
        v8f sc[4];
        #pragma unroll
        for (int nb = 0; nb < 4; ++nb)
            #pragma unroll
            for (int r = 0; r < 8; ++r) sc[nb][r] = 0.0f;
        __builtin_amdgcn_sched_barrier(0);
        #pragma unroll
        for (int nb = 0; nb < 4; ++nb) {
            sc[nb] = __builtin_amdgcn_wmma_f32_16x16x32_f16(
                false, a0, false, bfr[2 * nb], (short)0, sc[nb], false, false);
            sc[nb] = __builtin_amdgcn_wmma_f32_16x16x32_f16(
                false, a1, false, bfr[2 * nb + 1], (short)0, sc[nb], false, false);
        }

        // streaming softmax: row q = 8*half + r lives across 16 lanes of group
        #pragma unroll
        for (int r = 0; r < 8; ++r) {
            float tm = -1e30f;
            #pragma unroll
            for (int nb = 0; nb < 4; ++nb) tm = fmaxf(tm, sc[nb][r]);
            #pragma unroll
            for (int off = 1; off < 16; off <<= 1)
                tm = fmaxf(tm, __shfl_xor(tm, off, 32));
            tm *= scale;
            const float mnew = fmaxf(mrow[r], tm);
            const float corr = __expf(mrow[r] - mnew);
            float ssum = 0.f, osum = 0.f;
            #pragma unroll
            for (int nb = 0; nb < 4; ++nb) {
                const float e = __expf(sc[nb][r] * scale - mnew);
                ssum += e;
                osum += e * vs[cur][nb * 16 + lane16];
            }
            #pragma unroll
            for (int off = 1; off < 16; off <<= 1) {
                ssum += __shfl_xor(ssum, off, 32);
                osum += __shfl_xor(osum, off, 32);
            }
            mrow[r] = mnew;
            srow[r] = srow[r] * corr + ssum;
            orow[r] = orow[r] * corr + osum;
        }

        asm volatile("s_wait_asynccnt 0x0" ::: "memory");   // prefetch landed
        __syncthreads();                                     // all waves done w/ cur
    }

    if (lane16 == 0) {
        #pragma unroll
        for (int r = 0; r < 8; ++r) {
            const int q = qtile + wave * 16 + half * 8 + r;
            attn_out[((size_t)b * LQDIM + q) * NHEAD + h] = orow[r] / srow[r];
        }
    }
}

// ---------------------------------------------------------------------------
// Kernel 3: tiny MLP  y = relu(relu(o @ W1 + b1) @ W2 + b2), one thread/(b,q)
// ---------------------------------------------------------------------------
__global__ __launch_bounds__(256)
void mlp_kernel(const float* __restrict__ attn_out,
                const float* __restrict__ W1, const float* __restrict__ b1,
                const float* __restrict__ W2, const float* __restrict__ b2,
                float* __restrict__ y, int n)
{
    const int t = blockIdx.x * blockDim.x + threadIdx.x;
    if (t >= n) return;
    const float* o = attn_out + (size_t)t * NHEAD;
    float ov[NHEAD];
    #pragma unroll
    for (int i = 0; i < NHEAD; ++i) ov[i] = o[i];
    float acc2 = b2[0];
    #pragma unroll
    for (int j = 0; j < 10; ++j) {
        float hj = b1[j];
        #pragma unroll
        for (int i = 0; i < NHEAD; ++i) hj = fmaf(ov[i], W1[i * 10 + j], hj);
        hj = fmaxf(hj, 0.f);
        acc2 = fmaf(hj, W2[j], acc2);
    }
    y[t] = fmaxf(acc2, 0.f);
}

// ---------------------------------------------------------------------------
extern "C" void kernel_launch(void* const* d_in, const int* in_sizes, int n_in,
                              void* d_out, int out_size, void* d_ws, size_t ws_size,
                              hipStream_t stream)
{
    (void)in_sizes; (void)n_in; (void)out_size; (void)ws_size;
    const float* query = (const float*)d_in[0];
    const float* key   = (const float*)d_in[1];
    const float* value = (const float*)d_in[2];
    const float* Wq    = (const float*)d_in[3];
    const float* bq    = (const float*)d_in[4];
    const float* Wk    = (const float*)d_in[5];
    const float* bk    = (const float*)d_in[6];
    const float* W1    = (const float*)d_in[7];
    const float* b1    = (const float*)d_in[8];
    const float* W2    = (const float*)d_in[9];
    const float* b2    = (const float*)d_in[10];
    float* out = (float*)d_out;

    // workspace: qproj f16 [8,10,2048,64] | kproj f16 [8,10,1024,64] | attn f32
    const size_t qprojHalves = (size_t)BATCH * NHEAD * LQDIM * DPAD;
    const size_t kprojHalves = (size_t)BATCH * NHEAD * LKDIM * DPAD;
    _Float16* qproj = (_Float16*)d_ws;
    _Float16* kproj = qproj + qprojHalves;
    float*    attn  = (float*)(kproj + kprojHalves);          // ~32 MB total

    hipMemsetAsync(d_ws, 0, (qprojHalves + kprojHalves) * sizeof(_Float16), stream);

    const dim3 blk(128);
    proj_gemm_kernel<<<dim3((BATCH * LQDIM) / 64, 8), blk, 0, stream>>>(
        query, Wq, bq, qproj, QIN, LQDIM);
    proj_gemm_kernel<<<dim3((BATCH * LKDIM) / 64, 8), blk, 0, stream>>>(
        key, Wk, bk, kproj, KIN, LKDIM);
    attn_kernel<<<dim3(LQDIM / 64, BATCH * NHEAD), blk, 0, stream>>>(
        qproj, kproj, value, attn);
    const int n = BATCH * LQDIM;
    mlp_kernel<<<dim3((n + 255) / 256), dim3(256), 0, stream>>>(
        attn, W1, b1, W2, b2, out, n);
}
